// InContext_Predict_21766894256655
// MI455X (gfx1250) — matrix operand order repaired and checked by reading, not compiled, not run on hardware
//
#include <hip/hip_runtime.h>
#include <hip/hip_bf16.h>
#include <cstdint>
#include <cstddef>

// ---------------- problem dims (from reference) ----------------
constexpr int NX = 1024;      // query rows
constexpr int RR = 100000;    // reference rows  (1562*64 + 32)
constexpr int HH = 512;       // hidden dim (K of GEMM)
constexpr int OO = 256;       // output dim
constexpr int KNN = 1000;     // top-k

static_assert(RR % 64 == 32, "tail-strip assumption");
static_assert(RR % 16 == 0,  "tile assumption");

typedef _Float16 v16h __attribute__((ext_vector_type(16)));
typedef float    v8f  __attribute__((ext_vector_type(8)));
typedef _Float16 h4   __attribute__((ext_vector_type(4)));

// ---------------------------------------------------------------
// Kernel 1/2: row-normalize f32 rows and emit f16 rows.
// One wave (32 lanes) per row; lane handles 16 floats as 4x float4.
// ---------------------------------------------------------------
__global__ void normalize_rows_f16(const float* __restrict__ src,
                                   _Float16* __restrict__ dst,
                                   int nrows) {
    const int tid  = threadIdx.x;
    const int wid  = tid >> 5;
    const int lane = tid & 31;
    const int row  = blockIdx.x * 8 + wid;
    if (row >= nrows) return;

    const float* r = src + (size_t)row * HH;
    float4 v[4];
    float ss = 0.f;
#pragma unroll
    for (int j = 0; j < 4; ++j) {
        v[j] = *(const float4*)(r + j * 128 + lane * 4);
        ss += v[j].x * v[j].x + v[j].y * v[j].y + v[j].z * v[j].z + v[j].w * v[j].w;
    }
#pragma unroll
    for (int off = 16; off > 0; off >>= 1) ss += __shfl_xor(ss, off, 32);
    const float rinv = rsqrtf(ss);

    _Float16* d = dst + (size_t)row * HH;
#pragma unroll
    for (int j = 0; j < 4; ++j) {
        h4 h;
        h.x = (_Float16)(v[j].x * rinv);
        h.y = (_Float16)(v[j].y * rinv);
        h.z = (_Float16)(v[j].z * rinv);
        h.w = (_Float16)(v[j].w * rinv);
        *(h4*)(d + j * 128 + lane * 4) = h;
    }
}

// ---------------------------------------------------------------
// Branch-free strip GEMM: one wave computes a 16(M) x NT*16(N)
// strip of sim via v_wmma_f32_16x16x32_f16, k-loop over HH.
// A frag (16x32 f16, same k order as B): lane holds 16 contiguous halves
//   k = (lane>=16 ? 16..31 : 0..15) as two 16B loads, row M = lane&15.
// B frag (32x16 f16): lane holds 16 contiguous halves of column
//   N = lane&15, k = (lane>=16 ? 16..31 : 0..15) -> one 32B load.
// C layout: VGPR j, lanes 0-15 -> M=j; lanes 16-31 -> M=j+8.
// NO divergent control flow inside: accumulators stay in VGPRs,
// EXEC stays all-ones across every WMMA.
// ---------------------------------------------------------------
template <int NT>
__device__ __forceinline__ void gemm_strip(const _Float16* __restrict__ arow,
                                           const _Float16* __restrict__ rn16,
                                           _Float16* __restrict__ sim16,
                                           int tileM, int rbase, int lane) {
    const int koffA = (lane & 16) ? 16 : 0;
    const int koffB = (lane & 16) ? 16 : 0;
    const int l15   = lane & 15;
    const _Float16* brow = rn16 + (size_t)(rbase + l15) * HH + koffB;

    v8f c[NT];
#pragma unroll
    for (int t = 0; t < NT; ++t) c[t] = v8f{};

    for (int kk = 0; kk < HH; kk += 32) {
        union { uint4 u[2]; v16h v; } a;
        a.u[0] = *(const uint4*)(arow + kk + koffA);
        a.u[1] = *(const uint4*)(arow + kk + koffA + 8);
#pragma unroll
        for (int t = 0; t < NT; ++t) {
            const v16h b = *(const v16h*)(brow + (size_t)t * 16 * HH + kk);
            c[t] = __builtin_amdgcn_wmma_f32_16x16x32_f16(
                false, a.v, false, b, (short)0, c[t], false, false);
        }
    }

    const int mhi = (lane & 16) ? 8 : 0;
#pragma unroll
    for (int t = 0; t < NT; ++t) {
        const int rg = rbase + t * 16 + l15;
#pragma unroll
        for (int j = 0; j < 8; ++j) {
            sim16[(size_t)(tileM * 16 + mhi + j) * RR + rg] = (_Float16)c[t][j];
        }
    }
}

__global__ void sim_gemm_wmma(const _Float16* __restrict__ xn16,
                              const _Float16* __restrict__ rn16,
                              _Float16* __restrict__ sim16) {
    const int tid   = threadIdx.x;
    const int wid   = tid >> 5;
    const int lane  = tid & 31;
    const int tileM = blockIdx.y;                   // 0..63
    const int rbase = (blockIdx.x * 8 + wid) * 64;  // wave strip start
    if (rbase >= RR) return;

    const _Float16* arow = xn16 + (size_t)(tileM * 16 + (lane & 15)) * HH;

    // Wave-uniform choice: full 64-wide strip, or the single 32-wide tail.
    if (rbase + 64 <= RR) {
        gemm_strip<4>(arow, rn16, sim16, tileM, rbase, lane);
    } else {
        gemm_strip<2>(arow, rn16, sim16, tileM, rbase, lane);
    }
}

// ---------------------------------------------------------------
// Kernel 4: per query row — exact top-1000 by 2-level histogram
// select on f16 keys, softmax, fused weighted gather of ref_y.
// One block (256 threads) per row; output o == threadIdx.x.
// Tie-break matches jax.lax.top_k (lowest index first) via
// ballot-based ordered prefix over key==threshold elements.
// ---------------------------------------------------------------
__device__ __forceinline__ unsigned tkey(unsigned short s) {
    // order-preserving transform: bigger float -> bigger key
    return (s & 0x8000u) ? (unsigned)(unsigned short)(~s)
                         : (unsigned)(unsigned short)(s | 0x8000u);
}

__global__ void select_softmax_gather(const unsigned short* __restrict__ simu,
                                      const float* __restrict__ refy,
                                      float* __restrict__ pred) {
    __shared__ unsigned hist[256];
    __shared__ unsigned wtot[8];
    __shared__ int      rlist[256];
    __shared__ float    wlist[256];
    __shared__ unsigned maxtk, nsel, sBhi, sCgt, sTkT, sNeed;
    __shared__ float    sMax;

    const int tid  = threadIdx.x;
    const int wid  = tid >> 5;
    const int lane = tid & 31;
    const int n    = blockIdx.x;
    const unsigned short* srow = simu + (size_t)n * RR;

    // ---- Phase A: hi-byte histogram + row max ----
    hist[tid & 255] = 0;
    if (tid == 0) { maxtk = 0; nsel = 0; }
    __syncthreads();
    for (int r = tid; r < RR; r += 256) {
        const unsigned k = tkey(srow[r]);
        atomicAdd(&hist[k >> 8], 1u);
        atomicMax(&maxtk, k);
    }
    __syncthreads();
    if (tid == 0) {
        unsigned acc = 0, cgt = 0; int bh = 0;
        for (int b = 255; b >= 0; --b) {
            const unsigned h = hist[b];
            if (acc + h >= (unsigned)KNN) { bh = b; cgt = acc; break; }
            acc += h;
        }
        sBhi = (unsigned)bh; sCgt = cgt;
        const unsigned short t = (unsigned short)maxtk;
        union { unsigned short u; _Float16 h; } cv;
        cv.u = (t & 0x8000u) ? (unsigned short)(t & 0x7fffu) : (unsigned short)(~t);
        sMax = (float)cv.h;
    }
    __syncthreads();
    const unsigned bhi = sBhi;

    // ---- Phase B: low-byte histogram inside boundary bucket ----
    hist[tid & 255] = 0;
    __syncthreads();
    for (int r = tid; r < RR; r += 256) {
        const unsigned k = tkey(srow[r]);
        if ((k >> 8) == bhi) atomicAdd(&hist[k & 255u], 1u);
    }
    __syncthreads();
    if (tid == 0) {
        unsigned acc = sCgt;
        for (int b = 255; b >= 0; --b) {
            const unsigned h = hist[b];
            if (acc + h >= (unsigned)KNN) {
                sTkT  = (bhi << 8) | (unsigned)b;
                sNeed = (unsigned)KNN - acc;
                break;
            }
            acc += h;
        }
    }
    __syncthreads();
    const unsigned tkT  = sTkT;
    const unsigned need = sNeed;
    const float    mx   = sMax;

    // ---- Phase C: ordered streaming select + softmax-gather ----
    float acc = 0.f, wsum = 0.f;
    unsigned eq_base = 0;
    for (int base = 0; base < RR; base += 256) {
        const int r = base + tid;
        const bool inb = r < RR;
        const unsigned short raw = inb ? srow[r] : (unsigned short)0;
        const unsigned key = inb ? tkey(raw) : 0u;

        const bool eq = (key == tkT);
        const bool gt = (key > tkT);
        const unsigned long long bal = __ballot(eq);
        if (lane == 0) wtot[wid] = (unsigned)__popcll(bal);
        __syncthreads();
        const unsigned pre = (unsigned)__popcll(bal & ((1ull << lane) - 1ull));
        unsigned cbase = 0, ctot = 0;
#pragma unroll
        for (int w = 0; w < 8; ++w) {
            const unsigned tct = wtot[w];
            if (w < wid) cbase += tct;
            ctot += tct;
        }
        const bool sel = gt || (eq && (eq_base + cbase + pre) < need);
        if (sel) {
            const unsigned slot = atomicAdd(&nsel, 1u);
            union { unsigned short u; _Float16 h; } cv; cv.u = raw;
            rlist[slot] = r;
            wlist[slot] = __expf((float)cv.h - mx);
        }
        __syncthreads();
        const unsigned ns = nsel;
        __syncthreads();
        if (tid == 0) nsel = 0;
        for (unsigned i = 0; i < ns; ++i) {
            const float w = wlist[i];
            acc  += w * refy[(size_t)rlist[i] * OO + tid];
            wsum += w;
        }
        eq_base += ctot;
        __syncthreads();
    }
    pred[(size_t)n * OO + tid] = acc / wsum;
}

// ---------------------------------------------------------------
// Launch
// ---------------------------------------------------------------
extern "C" void kernel_launch(void* const* d_in, const int* in_sizes, int n_in,
                              void* d_out, int out_size, void* d_ws, size_t ws_size,
                              hipStream_t stream) {
    (void)in_sizes; (void)n_in; (void)out_size; (void)ws_size;
    const float* x     = (const float*)d_in[0];   // [1024, 512]
    const float* ref_x = (const float*)d_in[1];   // [100000, 512]
    const float* ref_y = (const float*)d_in[2];   // [100000, 256]
    float* out = (float*)d_out;                   // [1024, 256]

    char* ws = (char*)d_ws;
    _Float16* rn16  = (_Float16*)ws;                                      // 102,400,000 B
    _Float16* xn16  = (_Float16*)(ws + (size_t)RR * HH * 2);              //   1,048,576 B
    _Float16* sim16 = (_Float16*)(ws + (size_t)RR * HH * 2
                                     + (size_t)NX * HH * 2);              // 204,800,000 B

    normalize_rows_f16<<<RR / 8, 256, 0, stream>>>(ref_x, rn16, RR);
    normalize_rows_f16<<<NX / 8, 256, 0, stream>>>(x, xn16, NX);

    dim3 g((RR + 511) / 512, NX / 16);   // 196 x 64 blocks, 8 waves each
    sim_gemm_wmma<<<g, 256, 0, stream>>>(xn16, rn16, sim16);

    select_softmax_gather<<<NX, 256, 0, stream>>>(
        (const unsigned short*)sim16, ref_y, out);
}
